// CondConv2d_34505767256645
// MI455X (gfx1250) — compile-verified, matrix operand order checked by reading, not run on hardware
//
#include <hip/hip_runtime.h>

#define B_     16
#define CIN    256
#define COUT   256
#define H_     64
#define W_     64
#define HW     (H_*W_)
#define KP     9

#define BM 128
#define BN 256
#define BK 32
#define PITCH 40       // BK + 8 pad: rows stay 16B-aligned, avoids bank conflicts
#define NSTEP (KP * (CIN / BK))   // 72 K-steps

typedef __attribute__((ext_vector_type(8)))  _Float16 v8h;
typedef __attribute__((ext_vector_type(16))) _Float16 v16h;
typedef __attribute__((ext_vector_type(8)))  float    v8f;

// ---------------------------------------------------------------------------
// x [B][C][H][W] f32  ->  xh [B][H][W][C] f16  (NCHW->NHWC via LDS tile)
// ---------------------------------------------------------------------------
__global__ void cc_cvt_nhwc(const float* __restrict__ x,
                            _Float16* __restrict__ xh) {
    __shared__ float tile[64][65];
    const int cc = blockIdx.x;          // channel chunk of 64
    const int h  = blockIdx.y;
    const int b  = blockIdx.z;
    const int t  = threadIdx.x;
    const int w  = t & 63;
    const int q  = t >> 6;              // 0..3
    const float* src = x + (((size_t)b*CIN + cc*64)*H_ + h)*W_;
    #pragma unroll
    for (int i = 0; i < 16; ++i) {
        const int c = i*4 + q;
        tile[c][w] = src[(size_t)c*HW + w];          // coalesced along w
    }
    __syncthreads();
    const int c = t & 63;
    _Float16* dst = xh + (((size_t)b*H_ + h)*W_)*CIN + cc*64 + c;
    #pragma unroll
    for (int i = 0; i < 16; ++i) {
        const int ww = i*4 + q;
        dst[(size_t)ww*CIN] = (_Float16)tile[c][ww]; // coalesced along c
    }
}

// ---------------------------------------------------------------------------
// routing [B][8] x experts [8][COUT*CIN*9]  ->  wh [B][9][COUT][CIN] f16
// thread t = co*CIN + ci : reads 9 contiguous f32 per expert (coalesced),
// writes coalesced along ci for each kernel position.
// ---------------------------------------------------------------------------
__global__ void cc_combine(const float* __restrict__ routing,
                           const float* __restrict__ experts,
                           _Float16* __restrict__ wh) {
    const int b = blockIdx.y;
    const int t = blockIdx.x * 256 + threadIdx.x;    // < COUT*CIN
    float acc[9];
    #pragma unroll
    for (int k = 0; k < 9; ++k) acc[k] = 0.f;
    #pragma unroll
    for (int e = 0; e < 8; ++e) {
        const float r = routing[b*8 + e];            // scalar broadcast
        const float* p = experts + (size_t)e*(COUT*CIN*9) + (size_t)t*9;
        #pragma unroll
        for (int k = 0; k < 9; ++k) acc[k] = fmaf(r, p[k], acc[k]);
    }
    #pragma unroll
    for (int k = 0; k < 9; ++k)
        wh[((size_t)b*KP + k)*(COUT*CIN) + t] = (_Float16)acc[k];
}

// ---------------------------------------------------------------------------
// Implicit-GEMM conv, f16 WMMA, double-buffered LDS, 1 barrier / K-step.
// Per sample b: D[COUT,HW] = A[COUT,2304] x im2col(x); 72 K-steps of 32.
// Block: 256 thr = 8 waves (2M x 4N); block tile 128x256; wave tile 64x64.
// ---------------------------------------------------------------------------
struct Stage { v8h a0, a1, b0, b1, b2, b3; };

static __device__ __forceinline__
Stage load_step(int s, int b, int m0, int n0, int t,
                const _Float16* __restrict__ xh,
                const _Float16* __restrict__ wh) {
    Stage st;
    const int kp = s >> 3;              // 0..8 kernel position
    const int c0 = (s & 7) << 5;        // input-channel chunk
    const int dy = kp / 3 - 1;
    const int dx = kp % 3 - 1;
    // A: weights, 2 threads per row, 16 halves (32B) each
    const _Float16* wsrc = wh + (((size_t)b*KP + kp)*COUT + (m0 + (t >> 1)))*CIN
                              + ((t & 1) << 4) + c0;
    st.a0 = *reinterpret_cast<const v8h*>(wsrc);
    st.a1 = *reinterpret_cast<const v8h*>(wsrc + 8);
    // B: im2col of x (NHWC), 1 thread per row, 32 halves (64B)
    const int ng = n0 + t;
    const int py = (ng >> 6) + dy;
    const int px = (ng & 63) + dx;
    st.b0 = {}; st.b1 = {}; st.b2 = {}; st.b3 = {};
    if ((unsigned)py < (unsigned)H_ && (unsigned)px < (unsigned)W_) {
        const _Float16* xsrc = xh + (((size_t)b*H_ + py)*W_ + px)*CIN + c0;
        st.b0 = *reinterpret_cast<const v8h*>(xsrc);
        st.b1 = *reinterpret_cast<const v8h*>(xsrc + 8);
        st.b2 = *reinterpret_cast<const v8h*>(xsrc + 16);
        st.b3 = *reinterpret_cast<const v8h*>(xsrc + 24);
    }
    return st;
}

static __device__ __forceinline__
void store_stage(const Stage& st, _Float16* As, _Float16* Bs, int t) {
    const int sh = (t & 1) << 4;
    _Float16* ar = &As[(t >> 1) * PITCH + sh];
    *reinterpret_cast<v8h*>(ar    ) = st.a0;
    *reinterpret_cast<v8h*>(ar + 8) = st.a1;
    _Float16* br = &Bs[t * PITCH];
    *reinterpret_cast<v8h*>(br     ) = st.b0;
    *reinterpret_cast<v8h*>(br +  8) = st.b1;
    *reinterpret_cast<v8h*>(br + 16) = st.b2;
    *reinterpret_cast<v8h*>(br + 24) = st.b3;
}

__global__ __launch_bounds__(256)
void cc_conv_wmma(const _Float16* __restrict__ xh,
                  const _Float16* __restrict__ wh,
                  float* __restrict__ out) {
    __shared__ alignas(16) _Float16 As[2][BM * PITCH];   // [m][k]
    __shared__ alignas(16) _Float16 Bs[2][BN * PITCH];   // [n][k]

    const int b  = blockIdx.z;
    const int m0 = blockIdx.y * BM;
    const int n0 = blockIdx.x * BN;

    const int t    = threadIdx.x;
    const int lane = t & 31;
    const int wv   = t >> 5;            // 0..7
    const int wm   = wv >> 2;           // 0..1 -> 64 M rows
    const int wn   = wv & 3;            // 0..3 -> 64 N cols

    const int row = lane & 15;
    const int klo = (lane >> 4) << 3;   // 0 or 8 (16-bit A/B fragment layout)

    v8f acc[4][4] = {};

    // prologue: stage step 0 into buffer 0
    Stage st = load_step(0, b, m0, n0, t, xh, wh);
    store_stage(st, As[0], Bs[0], t);
    __syncthreads();

    for (int s = 0; s < NSTEP; ++s) {
        const int cur  = s & 1;
        const bool more = (s + 1) < NSTEP;
        Stage nxt;
        if (more) nxt = load_step(s + 1, b, m0, n0, t, xh, wh);  // overlap

        // ---- fragments: two ds_load_b128 per 16x32 fragment ----
        const _Float16* Ab = As[cur];
        const _Float16* Bb = Bs[cur];
        v16h afrag[4], bfrag[4];
        #pragma unroll
        for (int mi = 0; mi < 4; ++mi) {
            const int r = wm*64 + mi*16 + row;
            v8h lo = *reinterpret_cast<const v8h*>(&Ab[r*PITCH + klo]);
            v8h hi = *reinterpret_cast<const v8h*>(&Ab[r*PITCH + klo + 16]);
            afrag[mi] = __builtin_shufflevector(lo, hi,
                0,1,2,3,4,5,6,7,8,9,10,11,12,13,14,15);
        }
        #pragma unroll
        for (int ni = 0; ni < 4; ++ni) {
            const int r = wn*64 + ni*16 + row;
            v8h lo = *reinterpret_cast<const v8h*>(&Bb[r*PITCH + klo]);
            v8h hi = *reinterpret_cast<const v8h*>(&Bb[r*PITCH + klo + 16]);
            bfrag[ni] = __builtin_shufflevector(lo, hi,
                0,1,2,3,4,5,6,7,8,9,10,11,12,13,14,15);
        }

        // ---- 16 x v_wmma_f32_16x16x32_f16 ----
        #pragma unroll
        for (int mi = 0; mi < 4; ++mi)
            #pragma unroll
            for (int ni = 0; ni < 4; ++ni)
                acc[mi][ni] = __builtin_amdgcn_wmma_f32_16x16x32_f16(
                    false, afrag[mi], false, bfrag[ni],
                    (short)0, acc[mi][ni], false, false);

        // stage s+1 into the other buffer while this one is read-only
        if (more) store_stage(nxt, As[cur ^ 1], Bs[cur ^ 1], t);
        __syncthreads();
    }

    // ---- epilogue: C/D layout -> out[b][co][h][w] f32 ----
    const int col   = lane & 15;
    const int rbase = (lane >> 4) << 3;   // lanes 16-31 hold M = 8..15
    #pragma unroll
    for (int mi = 0; mi < 4; ++mi) {
        const int gm = m0 + wm*64 + mi*16 + rbase;
        #pragma unroll
        for (int ni = 0; ni < 4; ++ni) {
            const int gn = n0 + wn*64 + ni*16 + col;
            float* op = out + ((size_t)b*COUT + gm)*HW + gn;
            #pragma unroll
            for (int r = 0; r < 8; ++r)
                op[(size_t)r*HW] = acc[mi][ni][r];
        }
    }
}

// ---------------------------------------------------------------------------
extern "C" void kernel_launch(void* const* d_in, const int* in_sizes, int n_in,
                              void* d_out, int out_size, void* d_ws, size_t ws_size,
                              hipStream_t stream) {
    const float* x       = (const float*)d_in[0];
    const float* routing = (const float*)d_in[1];
    const float* experts = (const float*)d_in[2];
    float* out           = (float*)d_out;

    // workspace: xh (NHWC f16, 32 MiB) then wh ([b][9][co][ci] f16, 18 MiB)
    _Float16* xh = (_Float16*)d_ws;
    _Float16* wh = (_Float16*)((char*)d_ws + (size_t)B_ * HW * CIN * sizeof(_Float16));

    cc_cvt_nhwc <<<dim3(CIN/64, H_, B_),     256, 0, stream>>>(x, xh);
    cc_combine  <<<dim3((COUT*CIN)/256, B_), 256, 0, stream>>>(routing, experts, wh);
    cc_conv_wmma<<<dim3(HW/BN, COUT/BM, B_), 256, 0, stream>>>(xh, wh, out);
}
